// LocalAttentionBlock_74594991997394
// MI455X (gfx1250) — matrix operand order repaired and checked by reading, not compiled, not run on hardware
//
#include <hip/hip_runtime.h>
#include <hip/hip_bf16.h>

// ---------------------------------------------------------------------------
// MI455X (gfx1250) local-attention block: all matmuls via v_wmma_f32_16x16x32_bf16,
// attention K/V tiles staged via global_load_async_to_lds_b128 (ASYNCcnt).
// ---------------------------------------------------------------------------

typedef __attribute__((ext_vector_type(16))) __bf16 v16bf;
typedef __attribute__((ext_vector_type(8)))  __bf16 v8bf;
typedef __attribute__((ext_vector_type(8)))  float  v8f;

union BF16x16 { v16bf v; v8bf h[2]; };

#define TT   4096
#define WID  2560
#define NH   10
#define HD   256
#define WIN  2048
#define NQKV 3072   // 2560 (Q) + 256 (K) + 256 (V)

static __device__ __forceinline__ v8f wmma_bf16(v16bf a, v16bf b, v8f c) {
    return __builtin_amdgcn_wmma_f32_16x16x32_bf16(false, a, false, b, (short)0, c, false, false);
}

// Async global->LDS 16B copy (per-lane addresses), tracked by ASYNCcnt.
static __device__ __forceinline__ void async_load_b128(unsigned lds_off, unsigned long long gaddr) {
    asm volatile("global_load_async_to_lds_b128 %0, %1, off"
                 :: "v"(lds_off), "v"(gaddr) : "memory");
}
static __device__ __forceinline__ void wait_async0() {
#if __has_builtin(__builtin_amdgcn_s_wait_asynccnt)
    __builtin_amdgcn_s_wait_asynccnt(0);
#else
    asm volatile("s_wait_asynccnt 0x0" ::: "memory");
#endif
}

// ---------------------------------------------------------------------------
// Prep kernels
// ---------------------------------------------------------------------------
__global__ void cvt_x_kernel(const float* __restrict__ x, __bf16* __restrict__ xb, int n) {
    int i = blockIdx.x * 256 + threadIdx.x;
    if (i < n) xb[i] = (__bf16)x[i];
}

// Wt[n][k] = fused transposed QKV weight, n in [0,3072), k in [0,2560)
__global__ void build_wt_kernel(const float* __restrict__ Wq, const float* __restrict__ Wk,
                                const float* __restrict__ Wv, __bf16* __restrict__ Wt) {
    size_t i = (size_t)blockIdx.x * 256 + threadIdx.x;
    if (i >= (size_t)NQKV * WID) return;
    int n = (int)(i / WID);
    int k = (int)(i % WID);
    float v;
    if (n < WID)            v = Wq[(size_t)k * WID + n];
    else if (n < WID + HD)  v = Wk[(size_t)k * HD + (n - WID)];
    else                    v = Wv[(size_t)k * HD + (n - WID - HD)];
    Wt[i] = (__bf16)v;
}

__global__ void build_wot_kernel(const float* __restrict__ Wout, __bf16* __restrict__ WoT) {
    size_t i = (size_t)blockIdx.x * 256 + threadIdx.x;
    if (i >= (size_t)WID * WID) return;
    int n = (int)(i / WID);
    int k = (int)(i % WID);
    WoT[i] = (__bf16)Wout[(size_t)k * WID + n];
}

// ---------------------------------------------------------------------------
// Generic bf16 WMMA GEMM: C[M,N] = A[M,K] * Bt[N,K]^T (+ bias[n])
// One 16x16 tile per wave, 4 waves per block. K multiple of 32.
// All operands are L2-resident -> direct global b128 fragment loads.
// ---------------------------------------------------------------------------
__global__ __launch_bounds__(128)
void gemm_bf16_kernel(const __bf16* __restrict__ A, const __bf16* __restrict__ Bt,
                      const float* __restrict__ bias, float* __restrict__ C,
                      int M, int N, int K) {
    int wave = blockIdx.x * 4 + (threadIdx.x >> 5);
    int lane = threadIdx.x & 31;
    int tilesN = N >> 4;
    int tm = wave / tilesN;
    int tn = wave % tilesN;
    if (tm >= (M >> 4)) return;

    int r16 = lane & 15;
    int hi  = (lane >> 4) & 1;

    const __bf16* arow = A  + (size_t)(tm * 16 + r16) * K + hi * 8;   // A frag: K = b..b+7, b+16..b+23
    const __bf16* brow = Bt + (size_t)(tn * 16 + r16) * K + hi * 8;   // B frag: K = b..b+7, b+16..b+23

    v8f acc = {0.f, 0.f, 0.f, 0.f, 0.f, 0.f, 0.f, 0.f};
    for (int kb = 0; kb < K; kb += 32) {
        BF16x16 a, b;
        a.h[0] = *(const v8bf*)(arow + kb);
        a.h[1] = *(const v8bf*)(arow + kb + 16);
        b.h[0] = *(const v8bf*)(brow + kb);
        b.h[1] = *(const v8bf*)(brow + kb + 16);
        acc = wmma_bf16(a.v, b.v, acc);
    }

    float bval = bias ? bias[tn * 16 + r16] : 0.0f;
    float* crow = C + (size_t)(tm * 16 + hi * 8) * N + tn * 16 + r16;
#pragma unroll
    for (int r = 0; r < 8; ++r) crow[(size_t)r * N] = acc[r] + bval;
}

// ---------------------------------------------------------------------------
// RoPE: reads fp32 qkv[t][3072], writes bf16 Qb[t][2560] (pre-scaled by 1/16),
// Kb[t][256], Vb[t][256]. RoPE on first 128 dims of each head.
// ---------------------------------------------------------------------------
static __device__ __forceinline__ float rope_elem(const float* base, int hd, float pos) {
    if (hd < 64) {
        float ang = pos * __powf(10000.0f, -(float)hd * (1.0f / 64.0f));
        return base[hd] * __cosf(ang) - base[hd + 64] * __sinf(ang);
    } else if (hd < 128) {
        int j = hd - 64;
        float ang = pos * __powf(10000.0f, -(float)j * (1.0f / 64.0f));
        return base[hd] * __cosf(ang) + base[hd - 64] * __sinf(ang);
    }
    return base[hd];
}

__global__ void rope_kernel(const float* __restrict__ qkv, const int* __restrict__ segpos,
                            __bf16* __restrict__ Qb, __bf16* __restrict__ Kb,
                            __bf16* __restrict__ Vb) {
    int t   = blockIdx.x;
    int tid = threadIdx.x;
    float pos = (float)segpos[t];
    const float* row = qkv + (size_t)t * NQKV;

    for (int i = tid; i < WID; i += 256) {          // Q (10 heads)
        int hd = i & 255;
        float v = rope_elem(row + (i & ~255), hd, pos);
        Qb[(size_t)t * WID + i] = (__bf16)(v * 0.0625f);   // 1/sqrt(256)
    }
    if (tid < HD) {                                  // K (1 head)
        float v = rope_elem(row + WID, tid, pos);
        Kb[(size_t)t * HD + tid] = (__bf16)v;
    }
    if (tid < HD) {                                  // V passthrough
        Vb[(size_t)t * HD + tid] = (__bf16)row[WID + HD + tid];
    }
}

// ---------------------------------------------------------------------------
// Flash attention (sliding window 2048, causal). Block = 4 waves x 16 query
// rows; K and raw-V key tiles (16 KB each, contiguous) staged via async
// global->LDS b128 DMA, V transposed LDS->LDS. Per key tile per wave:
// 16 WMMAs for S = Q K^T, 16 WMMAs for O += P V. Online softmax with
// shfl_xor reductions over the 16-lane halves.
// ---------------------------------------------------------------------------
__global__ __launch_bounds__(128)
void attn_kernel(const __bf16* __restrict__ Qb, const __bf16* __restrict__ Kb,
                 const __bf16* __restrict__ Vb, __bf16* __restrict__ enc) {
    __shared__ __attribute__((aligned(16))) __bf16 sK [32 * HD];   // 16 KB: [key][256]
    __shared__ __attribute__((aligned(16))) __bf16 sVr[32 * HD];   // 16 KB: [key][256] raw
    __shared__ __attribute__((aligned(16))) __bf16 sVt[HD * 32];   // 16 KB: [dim][key]
    __shared__ __attribute__((aligned(16))) __bf16 sP [4][16 * 32];// 4 KB per-wave P tile

    const int tid  = threadIdx.x;
    const int w    = tid >> 5;
    const int lane = tid & 31;
    const int h    = blockIdx.y;
    const int q0   = blockIdx.x * 64;
    const int qw   = q0 + w * 16;
    const int r16  = lane & 15;
    const int hi   = (lane >> 4) & 1;
    const int koff = hi * 8;

    // Q fragments (row l%16, K-chunks per A-layout), held in VGPRs for all iters
    BF16x16 qf[8];
    {
        const __bf16* qrow = Qb + (size_t)(qw + r16) * WID + h * HD + hi * 8;
#pragma unroll
        for (int cb = 0; cb < 8; ++cb) {
            qf[cb].h[0] = *(const v8bf*)(qrow + cb * 32);
            qf[cb].h[1] = *(const v8bf*)(qrow + cb * 32 + 16);
        }
    }

    v8f o[16];
    const v8f vzero = {0.f, 0.f, 0.f, 0.f, 0.f, 0.f, 0.f, 0.f};
#pragma unroll
    for (int j = 0; j < 16; ++j) o[j] = vzero;
    float rowmax[8], rowsum[8];
#pragma unroll
    for (int r = 0; r < 8; ++r) { rowmax[r] = -1e30f; rowsum[r] = 0.0f; }

    const unsigned lK = (unsigned)(uintptr_t)(void*)sK;
    const unsigned lV = (unsigned)(uintptr_t)(void*)sVr;

    int kstart = q0 - WIN; if (kstart < 0) kstart = 0;

    for (int kb = kstart; kb < q0 + 64; kb += 32) {
        __syncthreads();  // previous iteration's LDS reads complete
        // async-stage K and raw V tiles (each 32x256 bf16 = 16 KB contiguous)
        {
            unsigned long long gK = (unsigned long long)(uintptr_t)(Kb + (size_t)kb * HD);
            unsigned long long gV = (unsigned long long)(uintptr_t)(Vb + (size_t)kb * HD);
            for (int c = tid; c < 1024; c += 128) {
                async_load_b128(lK + c * 16, gK + (unsigned long long)c * 16);
                async_load_b128(lV + c * 16, gV + (unsigned long long)c * 16);
            }
            wait_async0();
        }
        __syncthreads();
        // transpose V tile in LDS -> [256][32]
        for (int i2 = tid; i2 < 32 * HD; i2 += 128) {
            int k = i2 >> 8, n = i2 & 255;
            sVt[n * 32 + k] = sVr[k * HD + n];
        }
        __syncthreads();

        // S = Q K^T : two 16x16 key tiles, K-dim = 256
        v8f s0 = vzero, s1 = vzero;
#pragma unroll
        for (int cb = 0; cb < 8; ++cb) {
            BF16x16 b0, b1;
            const __bf16* k0 = &sK[r16 * HD + cb * 32 + koff];
            b0.h[0] = *(const v8bf*)(k0);
            b0.h[1] = *(const v8bf*)(k0 + 16);
            const __bf16* k1 = &sK[(16 + r16) * HD + cb * 32 + koff];
            b1.h[0] = *(const v8bf*)(k1);
            b1.h[1] = *(const v8bf*)(k1 + 16);
            s0 = wmma_bf16(qf[cb].v, b0.v, s0);
            s1 = wmma_bf16(qf[cb].v, b1.v, s1);
        }

        // online softmax: C-layout row m = r + hi*8, key col = kb(+16) + l%16
        const int j0 = kb + r16;
        const int j1 = kb + 16 + r16;
#pragma unroll
        for (int r = 0; r < 8; ++r) {
            int m = r + hi * 8;
            int qi = qw + m;
            float v0 = (j0 <= qi && j0 >= qi - WIN) ? s0[r] : -1e30f;
            float v1 = (j1 <= qi && j1 >= qi - WIN) ? s1[r] : -1e30f;
            float mr = fmaxf(rowmax[r], fmaxf(v0, v1));
            mr = fmaxf(mr, __shfl_xor(mr, 1, 32));
            mr = fmaxf(mr, __shfl_xor(mr, 2, 32));
            mr = fmaxf(mr, __shfl_xor(mr, 4, 32));
            mr = fmaxf(mr, __shfl_xor(mr, 8, 32));
            float scale = __expf(rowmax[r] - mr);
            float p0 = __expf(v0 - mr);
            float p1 = __expf(v1 - mr);
            float ps = p0 + p1;
            ps += __shfl_xor(ps, 1, 32);
            ps += __shfl_xor(ps, 2, 32);
            ps += __shfl_xor(ps, 4, 32);
            ps += __shfl_xor(ps, 8, 32);
            rowsum[r] = rowsum[r] * scale + ps;
            rowmax[r] = mr;
            sP[w][m * 32 + r16]      = (__bf16)p0;
            sP[w][m * 32 + 16 + r16] = (__bf16)p1;
#pragma unroll
            for (int j = 0; j < 16; ++j) o[j][r] *= scale;
        }
        // re-fragment P as A-matrix (DS ops are in-order within a wave)
        BF16x16 pa;
        {
            const __bf16* pp = &sP[w][r16 * 32 + hi * 8];
            pa.h[0] = *(const v8bf*)(pp);
            pa.h[1] = *(const v8bf*)(pp + 16);
        }
        // O += P V : 16 head-dim tiles, K-dim = 32 keys
#pragma unroll
        for (int j = 0; j < 16; ++j) {
            BF16x16 vb;
            const __bf16* vp = &sVt[(j * 16 + r16) * 32 + koff];
            vb.h[0] = *(const v8bf*)(vp);
            vb.h[1] = *(const v8bf*)(vp + 16);
            o[j] = wmma_bf16(pa.v, vb.v, o[j]);
        }
    }

    // normalize + store encoded bf16
#pragma unroll
    for (int r = 0; r < 8; ++r) {
        float inv = 1.0f / rowsum[r];
        int m = r + hi * 8;
        __bf16* erow = enc + (size_t)(qw + m) * WID + h * HD + r16;
#pragma unroll
        for (int j = 0; j < 16; ++j) erow[j * 16] = (__bf16)(o[j][r] * inv);
    }
}

// ---------------------------------------------------------------------------
// Host launcher
// ---------------------------------------------------------------------------
extern "C" void kernel_launch(void* const* d_in, const int* in_sizes, int n_in,
                              void* d_out, int out_size, void* d_ws, size_t ws_size,
                              hipStream_t stream) {
    (void)in_sizes; (void)n_in; (void)out_size; (void)ws_size;
    const float* x      = (const float*)d_in[0];
    const int*   segpos = (const int*)d_in[1];
    const float* Wq     = (const float*)d_in[2];
    const float* Wk     = (const float*)d_in[3];
    const float* Wv     = (const float*)d_in[4];
    const float* Wout   = (const float*)d_in[5];
    const float* b_out  = (const float*)d_in[6];
    float* out = (float*)d_out;

    // workspace layout (256B aligned regions, deterministic reuse)
    char* w = (char*)d_ws;
    size_t o = 0;
    auto take = [&](size_t n) { char* p = w + o; o += (n + 255) & ~(size_t)255; return p; };
    __bf16* xb  = (__bf16*)take((size_t)TT * WID * 2);    // x bf16; reused as Qb after rope
    __bf16* Wt  = (__bf16*)take((size_t)NQKV * WID * 2);  // fused transposed QKV weights
    __bf16* WoT = (__bf16*)take((size_t)WID * WID * 2);   // transposed Wout
    float*  qkv = (float*)take((size_t)TT * NQKV * 4);    // fp32 QKV; reused as encoded bf16
    __bf16* Kb  = (__bf16*)take((size_t)TT * HD * 2);
    __bf16* Vb  = (__bf16*)take((size_t)TT * HD * 2);
    __bf16* Qb  = xb;           // reuse: xb dead after QKV GEMM
    __bf16* enc = (__bf16*)qkv; // reuse: qkv dead after rope

    // 1. prep
    cvt_x_kernel<<<(TT * WID + 255) / 256, 256, 0, stream>>>(x, xb, TT * WID);
    build_wt_kernel<<<((size_t)NQKV * WID + 255) / 256, 256, 0, stream>>>(Wq, Wk, Wv, Wt);
    build_wot_kernel<<<((size_t)WID * WID + 255) / 256, 256, 0, stream>>>(Wout, WoT);

    // 2. fused QKV projection: [4096,2560] x [2560,3072] -> fp32
    {
        int tiles = (TT / 16) * (NQKV / 16);
        gemm_bf16_kernel<<<tiles / 4, 128, 0, stream>>>(xb, Wt, nullptr, qkv, TT, NQKV, WID);
    }

    // 3. RoPE + scale + bf16 split
    rope_kernel<<<TT, 256, 0, stream>>>(qkv, segpos, Qb, Kb, Vb);

    // 4. sliding-window flash attention
    attn_kernel<<<dim3(TT / 64, NH), 128, 0, stream>>>(Qb, Kb, Vb, enc);

    // 5. output projection with bias
    {
        int tiles = (TT / 16) * (WID / 16);
        gemm_bf16_kernel<<<tiles / 4, 128, 0, stream>>>(enc, WoT, b_out, out, TT, WID, WID);
    }
}
